// SelfAttention_36730560315813
// MI455X (gfx1250) — matrix operand order repaired and checked by reading, not compiled, not run on hardware
//
#include <hip/hip_runtime.h>
#include <cstddef>
#include <cstdint>

typedef unsigned short u16;
typedef unsigned int   u32;
typedef unsigned long long u64;

typedef __attribute__((ext_vector_type(16))) __bf16 v16bf;
typedef __attribute__((ext_vector_type(2)))  __bf16 v2bf;
typedef __attribute__((ext_vector_type(8)))  float  v8f;

// LDS row pitch in bf16 elems: 32 data + 24 pad = 112B rows.
// 112B is a multiple of 16B (b128-aligned for every row) and 28 banks,
// so 16 consecutive rows map to 16 distinct bank groups (conflict-free).
#define LDSP 56

__device__ __forceinline__ u16 bf_bits(float f) {
  __bf16 h = (__bf16)f;                  // hardware RNE f32->bf16
  return __builtin_bit_cast(u16, h);
}

union FragU { v16bf v; u32 u[8]; };

// A-matrix 16x32 bf16 fragment per ISA table: lanes 0-15 row M=lane, K in {0..7,16..23};
// lanes 16-31 row M=lane-16, K in {8..15,24..31}. Two bf16 per VGPR, K-contiguous pairs.
__device__ __forceinline__ v16bf load_frag_a(const u16* base, int lane) {
  const int lr  = lane & 15;
  const int kb2 = (lane >> 4) * 4;  // u32 index base: K=0 -> 0, K=8 -> 4
  const u32* src = (const u32*)(base + lr * LDSP);
  FragU f;
#pragma unroll
  for (int v = 0; v < 4; ++v) f.u[v] = src[kb2 + v];
#pragma unroll
  for (int v = 0; v < 4; ++v) f.u[4 + v] = src[kb2 + 8 + v];
  return f.v;
}

// B-matrix 32x16 bf16 fragment: lane = column N; lanes 0-15 hold K in {0..7,16..23},
// lanes 16-31 hold K in {8..15,24..31} (LDS tile holds B^T rows, i.e. K-contiguous per column).
__device__ __forceinline__ v16bf load_frag_b(const u16* base, int lane) {
  const int lr   = lane & 15;
  const int koff = (lane >> 4) * 4;  // u32 index base: 0 or 4
  const u32* src = (const u32*)(base + lr * LDSP);
  FragU f;
#pragma unroll
  for (int v = 0; v < 8; ++v) f.u[v] = src[koff + (v < 4 ? v : v + 4)];
  return f.v;
}

// Async 16B copy global -> LDS (CDNA5 GLOBAL_LOAD_ASYNC_TO_LDS_B128, ASYNCcnt-tracked).
__device__ __forceinline__ void async_copy_b128(const u16* gptr, const u16* lds_ptr) {
  u32 lp = (u32)(uintptr_t)lds_ptr;     // low 32 bits = LDS-relative offset
  u64 gp = (u64)(uintptr_t)gptr;
  asm volatile("global_load_async_to_lds_b128 %0, %1, off"
               :: "v"(lp), "v"(gp) : "memory");
}

__device__ __forceinline__ void async_wait0() {
  asm volatile("s_wait_asynccnt 0x0" ::: "memory");
}

enum { MODE_F32 = 0, MODE_F32_BIAS = 1, MODE_BF16 = 2, MODE_QKV = 3 };

// C[M,N] = A[M,K] * B[K,N], with B provided pre-transposed as BT[N,K] (K-contiguous).
// Block tile 128x128, BK=32, 8 waves; each wave: 2x4 WMMA tiles (32 rows x 64 cols).
template<int MODE, bool AF32>
__global__ __launch_bounds__(256) void gemm_wmma(
    const void* __restrict__ Av, int lda,
    const u16* __restrict__ BT, int ldb, int K,
    float* __restrict__ Cf, u16* __restrict__ Cb, int ldc,
    const float* __restrict__ bias,
    u16* __restrict__ q_out, u16* __restrict__ k_out, u16* __restrict__ vt_out)
{
  __shared__ u16 As[128 * LDSP];
  __shared__ u16 Bs[128 * LDSP];

  const int tid  = threadIdx.x;
  const int lane = tid & 31;
  const int wave = tid >> 5;
  const int wm = wave & 3;   // 4 row groups of 32
  const int wn = wave >> 2;  // 2 col groups of 64
  const int blockRow = blockIdx.y * 128;
  const int blockCol = blockIdx.x * 128;

  v8f acc[2][4] = {};

  for (int k0 = 0; k0 < K; k0 += 32) {
    __syncthreads();
    // ---- A tile (128 x 32) -> LDS ----
    if constexpr (AF32) {
      // f32 source: batch the 4 global loads (overlapped), then cvt+store.
      const float* A = (const float*)Av;
      float4 fv[4];
#pragma unroll
      for (int i = 0; i < 4; ++i) {
        int linear = tid + i * 256;        // 1024 float4 units
        int r  = linear >> 3;
        int c4 = linear & 7;
        fv[i] = *(const float4*)(A + (size_t)(blockRow + r) * lda + k0 + c4 * 4);
      }
#pragma unroll
      for (int i = 0; i < 4; ++i) {
        int linear = tid + i * 256;
        int r  = linear >> 3;
        int c4 = linear & 7;
        v2bf* dst = (v2bf*)&As[r * LDSP + c4 * 4];
        v2bf p0; p0.x = (__bf16)fv[i].x; p0.y = (__bf16)fv[i].y;
        v2bf p1; p1.x = (__bf16)fv[i].z; p1.y = (__bf16)fv[i].w;
        dst[0] = p0; dst[1] = p1;
      }
    } else {
      // bf16 source: async DMA straight into LDS (no VGPR round-trip).
      const u16* A = (const u16*)Av;
#pragma unroll
      for (int i = 0; i < 2; ++i) {
        int linear = tid + i * 256;        // 512 x 16B chunks
        int r = linear >> 2;
        int c = linear & 3;
        async_copy_b128(A + (size_t)(blockRow + r) * lda + k0 + c * 8,
                        &As[r * LDSP + c * 8]);
      }
    }
    // ---- BT tile (128 x 32) -> LDS via async DMA ----
#pragma unroll
    for (int i = 0; i < 2; ++i) {
      int linear = tid + i * 256;
      int r = linear >> 2;
      int c = linear & 3;
      async_copy_b128(BT + (size_t)(blockCol + r) * ldb + k0 + c * 8,
                      &Bs[r * LDSP + c * 8]);
    }
    // prefetch next K slab of BT (lowers to global_prefetch_b8)
    if (k0 + 32 < K) {
      int r = tid >> 1;
      __builtin_prefetch(BT + (size_t)(blockCol + r) * ldb + k0 + 32, 0, 1);
    }
    async_wait0();        // ASYNCcnt: our async tile copies have landed in LDS
    __syncthreads();      // (compiler adds the DScnt wait for the cvt path)

    v16bf afr[2], bfr[4];
#pragma unroll
    for (int i = 0; i < 2; ++i) afr[i] = load_frag_a(&As[(wm * 32 + i * 16) * LDSP], lane);
#pragma unroll
    for (int j = 0; j < 4; ++j) bfr[j] = load_frag_b(&Bs[(wn * 64 + j * 16) * LDSP], lane);
#pragma unroll
    for (int i = 0; i < 2; ++i)
#pragma unroll
      for (int j = 0; j < 4; ++j)
        acc[i][j] = __builtin_amdgcn_wmma_f32_16x16x32_bf16(
            false, afr[i], false, bfr[j], (short)0, acc[i][j], false, false);
  }

  // ---- epilogue: C/D layout: VGPR v -> row v (+8 for lanes 16-31), lane&15 -> col ----
  const int mbase = (lane >> 4) * 8;
  const int nloc  = lane & 15;
#pragma unroll
  for (int i = 0; i < 2; ++i) {
#pragma unroll
    for (int j = 0; j < 4; ++j) {
      const int tm = blockRow + wm * 32 + i * 16;
      const int gn = blockCol + wn * 64 + j * 16 + nloc;
#pragma unroll
      for (int v = 0; v < 8; ++v) {
        const int gm = tm + mbase + v;
        float val = acc[i][j][v];
        if constexpr (MODE == MODE_F32) {
          Cf[(size_t)gm * ldc + gn] = val;
        } else if constexpr (MODE == MODE_F32_BIAS) {
          Cf[(size_t)gm * ldc + gn] = val + bias[gn];
        } else if constexpr (MODE == MODE_BF16) {
          Cb[(size_t)gm * ldc + gn] = bf_bits(val);
        } else {  // MODE_QKV: split into Q (scaled), K, V^T; bias over full 3072
          val += bias[gn];
          const int region = gn >> 10;
          const int c = gn & 1023;
          if (region == 0) {
            q_out[(size_t)gm * 1024 + c] = bf_bits(val * 0.125f);   // fold 1/sqrt(64)
          } else if (region == 1) {
            k_out[(size_t)gm * 1024 + c] = bf_bits(val);
          } else {
            const int b = gm >> 11;          // gm = b*2048 + s
            const int s = gm & 2047;
            vt_out[((size_t)b << 21) + (size_t)c * 2048 + s] = bf_bits(val);
          }
        }
      }
    }
  }
}

// Row softmax over [2048] f32, written back in place as bf16 (P rows at pitch 4096 u16).
__global__ __launch_bounds__(256) void softmax_inplace(float* __restrict__ S) {
  __shared__ float red[256];
  const int tid = threadIdx.x;
  float* rp = S + (size_t)blockIdx.x * 2048;
  float v[8];
  float m = -3.4e38f;
#pragma unroll
  for (int i = 0; i < 8; ++i) { v[i] = rp[tid + (i << 8)]; m = fmaxf(m, v[i]); }
  red[tid] = m; __syncthreads();
  for (int s = 128; s > 0; s >>= 1) { if (tid < s) red[tid] = fmaxf(red[tid], red[tid + s]); __syncthreads(); }
  m = red[0]; __syncthreads();
  float sum = 0.f;
#pragma unroll
  for (int i = 0; i < 8; ++i) { v[i] = __expf(v[i] - m); sum += v[i]; }
  red[tid] = sum; __syncthreads();
  for (int s = 128; s > 0; s >>= 1) { if (tid < s) red[tid] += red[tid + s]; __syncthreads(); }
  const float inv = 1.0f / red[0];
  u16* o = (u16*)rp;
#pragma unroll
  for (int i = 0; i < 8; ++i) o[tid + (i << 8)] = bf_bits(v[i] * inv);
}

// out[C][R] = bf16(in[R][C])
__global__ __launch_bounds__(256) void transpose_to_bf16(const float* __restrict__ in,
                                                         u16* __restrict__ out, int R, int C) {
  size_t i = (size_t)blockIdx.x * 256 + threadIdx.x;
  int r = (int)(i / C);
  int c = (int)(i % C);
  out[(size_t)c * R + r] = bf_bits(in[i]);
}

extern "C" void kernel_launch(void* const* d_in, const int* in_sizes, int n_in,
                              void* d_out, int out_size, void* d_ws, size_t ws_size,
                              hipStream_t stream) {
  (void)in_sizes; (void)n_in; (void)out_size; (void)ws_size;
  const float* x     = (const float*)d_in[0];   // [4,2048,1024]
  const float* W_qkv = (const float*)d_in[1];   // [1024,3072]
  const float* b_qkv = (const float*)d_in[2];   // [3072]
  const float* W_out = (const float*)d_in[3];   // [1024,1024]
  const float* b_out = (const float*)d_in[4];   // [1024]
  float* out = (float*)d_out;                   // [4,2048,1024] f32

  char* ws = (char*)d_ws;
  u16* WqkvT = (u16*)ws;  ws += (size_t)3072 * 1024 * 2;   // [3072,1024] bf16
  u16* WoutT = (u16*)ws;  ws += (size_t)1024 * 1024 * 2;   // [1024,1024] bf16
  u16* Qb    = (u16*)ws;  ws += (size_t)8192 * 1024 * 2;   // [B*S,1024] bf16 (pre-scaled)
  u16* Kb    = (u16*)ws;  ws += (size_t)8192 * 1024 * 2;   // [B*S,1024] bf16
  u16* VTb   = (u16*)ws;  ws += (size_t)8192 * 1024 * 2;   // [B][1024][2048] bf16 (V^T)
  u16* CTX   = (u16*)ws;  ws += (size_t)8192 * 1024 * 2;   // [B*S,1024] bf16
  float* SC  = (float*)ws;                                  // [2048,2048] f32, reused per batch

  // weight transposes (bf16, K-contiguous B operands)
  transpose_to_bf16<<<dim3((1024 * 3072) / 256), 256, 0, stream>>>(W_qkv, WqkvT, 1024, 3072);
  transpose_to_bf16<<<dim3((1024 * 1024) / 256), 256, 0, stream>>>(W_out, WoutT, 1024, 1024);

  // fused QKV projection: [8192,1024] x [1024,3072] -> Q (scaled), K, V^T
  gemm_wmma<MODE_QKV, true><<<dim3(3072 / 128, 8192 / 128), 256, 0, stream>>>(
      x, 1024, WqkvT, 1024, 1024, nullptr, nullptr, 0, b_qkv, Qb, Kb, VTb);

  for (int b = 0; b < 4; ++b) {
    const u16* Qp = Qb + (size_t)b * 2048 * 1024;
    const u16* Kp = Kb + (size_t)b * 2048 * 1024;
    const u16* Vp = VTb + (size_t)b * 1024 * 2048;
    u16* Cp = CTX + (size_t)b * 2048 * 1024;
    // scores = Q @ K^T  (BT operand is K itself, [S,E] row-major)
    gemm_wmma<MODE_F32, false><<<dim3(2048 / 128, 2048 / 128), 256, 0, stream>>>(
        Qp, 1024, Kp, 1024, 1024, SC, nullptr, 2048, nullptr, nullptr, nullptr, nullptr);
    // softmax rows, in place f32 -> bf16 (P rows at u16 pitch 4096)
    softmax_inplace<<<dim3(2048), 256, 0, stream>>>(SC);
    // context = P @ V  (BT operand is V^T [1024,2048])
    gemm_wmma<MODE_BF16, false><<<dim3(1024 / 128, 2048 / 128), 256, 0, stream>>>(
        (const void*)SC, 4096, Vp, 2048, 2048, nullptr, Cp, 1024, nullptr, nullptr, nullptr, nullptr);
  }

  // output projection: [8192,1024] x [1024,1024] + bias -> f32 out
  gemm_wmma<MODE_F32_BIAS, false><<<dim3(1024 / 128, 8192 / 128), 256, 0, stream>>>(
      CTX, 1024, WoutT, 1024, 1024, out, nullptr, 1024, b_out, nullptr, nullptr, nullptr);
}
